// FlaxRoFormerSelfAttention_59150289601143
// MI455X (gfx1250) — compile-verified, matrix operand order checked by reading, not run on hardware
//
#include <hip/hip_runtime.h>
#include <stddef.h>
#include <stdint.h>

// ---------------------------------------------------------------------------
// RoFormer self-attention forward for MI455X (gfx1250, wave32, WMMA bf16)
// B=2, S=2048, D=1024, H=16, DH=64
// ---------------------------------------------------------------------------

typedef unsigned short u16;
typedef __attribute__((ext_vector_type(16))) __bf16       v16bf;
typedef __attribute__((ext_vector_type(8)))  float        v8f;
typedef __attribute__((ext_vector_type(4)))  unsigned int v4u;
typedef __attribute__((ext_vector_type(4)))  int          v4i;

union Frag {            // one 16x32 bf16 A/B fragment: 8 VGPRs per lane
    v16bf v;
    v4u   u[2];
};

#define WMMA_BF16(a, b, c) \
    __builtin_amdgcn_wmma_f32_16x16x32_bf16(false, (a), false, (b), (short)0, (c), false, false)

// ---- gfx1250 async global->LDS staging (ASYNCcnt path), with sync fallback --
#if defined(__has_builtin)
#if __has_builtin(__builtin_amdgcn_global_load_async_to_lds_b128)
#define HAVE_ASYNC_LDS 1
#endif
#endif
#ifndef HAVE_ASYNC_LDS
#define HAVE_ASYNC_LDS 0
#endif

#if HAVE_ASYNC_LDS
typedef __attribute__((address_space(1))) v4i gv4i;   // global int4
typedef __attribute__((address_space(3))) v4i lv4i;   // LDS int4
#endif

__device__ __forceinline__ void async_copy16(const u16* gsrc, u16* ldst) {
#if HAVE_ASYNC_LDS
    __builtin_amdgcn_global_load_async_to_lds_b128(
        (gv4i*)(uintptr_t)gsrc,
        (lv4i*)(unsigned int)(uintptr_t)ldst,
        0, 0);
#else
    *(v4u*)ldst = *(const v4u*)gsrc;
#endif
}

__device__ __forceinline__ void wait_async() {
#if HAVE_ASYNC_LDS
#if __has_builtin(__builtin_amdgcn_s_wait_asynccnt)
    __builtin_amdgcn_s_wait_asynccnt(0);
#else
    asm volatile("s_wait_asynccnt 0" ::: "memory");
#endif
#endif
}

__device__ __forceinline__ u16 f2bf(float x) {
    unsigned int u = __float_as_uint(x);
    unsigned int r = u + 0x7FFFu + ((u >> 16) & 1u);   // round-to-nearest-even
    return (u16)(r >> 16);
}

static constexpr int Bc = 2, Sc = 2048, Dc = 1024, Hc = 16, DHc = 64;

// ---------------------------------------------------------------------------
// Kernel 0a: fp32 -> bf16 convert (hidden states)
// ---------------------------------------------------------------------------
__global__ void cvt_bf16_kernel(const float* __restrict__ src, u16* __restrict__ dst, int n) {
    int i = blockIdx.x * blockDim.x + threadIdx.x;
    if (i < n) dst[i] = f2bf(src[i]);
}

// ---------------------------------------------------------------------------
// Kernel 0b: fp32 [K][N] -> bf16 transposed [N][K]  (weights; 32x32 LDS tiles)
// Pre-transposing W makes the GEMM B-staging identical to A-staging
// (pure b128 traffic, no scalar ds scatter in the hot loop).
// ---------------------------------------------------------------------------
__global__ void __launch_bounds__(256)
cvt_transpose_kernel(const float* __restrict__ src, u16* __restrict__ dst) {
    __shared__ u16 tile[32][33];
    const int bx = blockIdx.x * 32;              // N base
    const int by = blockIdx.y * 32;              // K base
    const int tx = threadIdx.x & 31, ty = threadIdx.x >> 5;   // 32 x 8
#pragma unroll
    for (int i = 0; i < 32; i += 8)              // read rows k coalesced over n
        tile[tx][ty + i] = f2bf(src[(size_t)(by + ty + i) * Dc + bx + tx]);
    __syncthreads();
#pragma unroll
    for (int i = 0; i < 32; i += 8)              // write rows n coalesced over k
        dst[(size_t)(bx + ty + i) * Dc + by + tx] = tile[ty + i][tx];
}

// ---------------------------------------------------------------------------
// Kernel 1: fused QKV projection GEMM  (z = 0/1/2 -> Q/K/V)
//   X[4096,1024] bf16  @  Wt[N=1024][K=1024] bf16 (pre-transposed)  + bias
//   block: 256 thr (8 waves), 128x128 tile, wave = 64x32 subtile,
//   double-buffered 32-deep K stages via async global->LDS.
// ---------------------------------------------------------------------------
__global__ void __launch_bounds__(256)
qkv_gemm_kernel(const u16* __restrict__ X,
                const u16* __restrict__ Wq, const u16* __restrict__ Wk, const u16* __restrict__ Wv,
                const float* __restrict__ bq, const float* __restrict__ bk, const float* __restrict__ bv,
                float* __restrict__ Qr, float* __restrict__ Kr, float* __restrict__ Vr) {
    const int z = blockIdx.z;
    const u16*   W    = (z == 0) ? Wq : (z == 1) ? Wk : Wv;   // [N][K]
    const float* bias = (z == 0) ? bq : (z == 1) ? bk : bv;
    float*       Out  = (z == 0) ? Qr : (z == 1) ? Kr : Vr;

    __shared__ __attribute__((aligned(16))) u16 lA[2][128][40];   // 80B stride: conflict-free
    __shared__ __attribute__((aligned(16))) u16 lB[2][128][40];   // [n][k] (W pre-transposed)

    const int t = threadIdx.x;
    const int lane = t & 31, wave = t >> 5;
    const int waveM = wave >> 2, waveN = wave & 3;       // 2 x 4 wave grid
    const int blockM = blockIdx.y * 128, blockN = blockIdx.x * 128;

    const int m16 = lane & 15;
    const int c   = (lane < 16) ? 0 : 8;                 // per-half K offset (ISA layout)
    const int lh  = lane >> 4;

    const int sr = t >> 1, sc = (t & 1) * 16;            // staging: 128 rows x 2 chunks

    auto stage = [&](int buf, int k0) {
        const u16* sa = X + (size_t)(blockM + sr) * Dc + k0 + sc;
        async_copy16(sa,     &lA[buf][sr][sc]);
        async_copy16(sa + 8, &lA[buf][sr][sc + 8]);
        const u16* sb = W + (size_t)(blockN + sr) * Dc + k0 + sc;
        async_copy16(sb,     &lB[buf][sr][sc]);
        async_copy16(sb + 8, &lB[buf][sr][sc + 8]);
    };

    v8f acc[4][2] = {};
    stage(0, 0);
    int cur = 0;

    for (int k0 = 0; k0 < Dc; k0 += 32) {
        wait_async();
        __syncthreads();                          // buf[cur] ready on all waves
        if (k0 + 32 < Dc) stage(cur ^ 1, k0 + 32);  // prefetch next stage

        Frag bf[2];
#pragma unroll
        for (int nt = 0; nt < 2; ++nt) {
            int n = waveN * 32 + nt * 16 + m16;
            bf[nt].u[0] = *(const v4u*)&lB[cur][n][c];
            bf[nt].u[1] = *(const v4u*)&lB[cur][n][c + 16];
        }
#pragma unroll
        for (int mt = 0; mt < 4; ++mt) {
            int m = waveM * 64 + mt * 16 + m16;
            Frag af;
            af.u[0] = *(const v4u*)&lA[cur][m][c];
            af.u[1] = *(const v4u*)&lA[cur][m][c + 16];
#pragma unroll
            for (int nt = 0; nt < 2; ++nt)
                acc[mt][nt] = WMMA_BF16(af.v, bf[nt].v, acc[mt][nt]);
        }
        cur ^= 1;
    }

    // epilogue: C layout is m = v + 8*lh, n = m16
#pragma unroll
    for (int mt = 0; mt < 4; ++mt)
#pragma unroll
        for (int nt = 0; nt < 2; ++nt) {
            int gn = blockN + waveN * 32 + nt * 16 + m16;
            float bsv = bias[gn];
#pragma unroll
            for (int v = 0; v < 8; ++v) {
                int gm = blockM + waveM * 64 + mt * 16 + v + 8 * lh;
                Out[(size_t)gm * Dc + gn] = acc[mt][nt][v] + bsv;
            }
        }
}

// ---------------------------------------------------------------------------
// Kernel 2: RoPE + repack to attention-friendly bf16 layouts
//   qh,kh : [b,h,s,dh]  (q scaled by 1/sqrt(64))
//   vt    : [b,h,dh,s]  (transposed so P@V B-fragments are contiguous)
// ---------------------------------------------------------------------------
__global__ void rope_pack_kernel(const float* __restrict__ Qr, const float* __restrict__ Kr,
                                 const float* __restrict__ Vr, const float* __restrict__ sp,
                                 u16* __restrict__ qh, u16* __restrict__ kh, u16* __restrict__ vt) {
    int idx = blockIdx.x * blockDim.x + threadIdx.x;   // B*S*H*32 threads
    int i   = idx & 31;  idx >>= 5;
    int h   = idx & 15;  idx >>= 4;
    int s   = idx & 2047;
    int b   = idx >> 11;

    size_t r = (size_t)(b * Sc + s) * Dc + h * DHc + 2 * i;
    float q0 = Qr[r], q1 = Qr[r + 1];
    float k0 = Kr[r], k1 = Kr[r + 1];
    float v0 = Vr[r], v1 = Vr[r + 1];
    float sn = sp[s * DHc + i];
    float cs = sp[s * DHc + 32 + i];

    float q0r = (q0 * cs - q1 * sn) * 0.125f;
    float q1r = (q1 * cs + q0 * sn) * 0.125f;
    float k0r =  k0 * cs - k1 * sn;
    float k1r =  k1 * cs + k0 * sn;

    size_t o = ((size_t)(b * Hc + h) * Sc + s) * DHc + 2 * i;
    qh[o] = f2bf(q0r);  qh[o + 1] = f2bf(q1r);
    kh[o] = f2bf(k0r);  kh[o + 1] = f2bf(k1r);

    size_t ov = ((size_t)(b * Hc + h) * DHc + 2 * i) * Sc + s;
    vt[ov] = f2bf(v0);  vt[ov + Sc] = f2bf(v1);
}

// ---------------------------------------------------------------------------
// Kernel 3: flash-style attention. 1 block = (qtile of 64, h, b), 4 waves.
// Each wave owns 16 q rows, streams 64-key chunks.
// ---------------------------------------------------------------------------
__global__ void __launch_bounds__(128)
attn_kernel(const u16* __restrict__ qh, const u16* __restrict__ kh, const u16* __restrict__ vt,
            const int* __restrict__ mask, const float* __restrict__ hm, float* __restrict__ out) {
    const int b = blockIdx.z, h = blockIdx.y, qt = blockIdx.x;
    const int t = threadIdx.x;
    const int lane = t & 31, w = t >> 5;
    const size_t bh = (size_t)(b * Hc + h);

    const int m16 = lane & 15;
    const int c   = (lane < 16) ? 0 : 8;
    const int lh  = lane >> 4;

    __shared__ __attribute__((aligned(16))) u16 pP[4][16][72];  // per-wave P transpose buffer

    // Q fragments (A layout): row = q position, K-dim = dh
    const int qrow = qt * 64 + w * 16 + m16;
    const u16* qb = qh + (bh * Sc + qrow) * DHc;
    Frag aq[2];
    aq[0].u[0] = *(const v4u*)(qb + c);
    aq[0].u[1] = *(const v4u*)(qb + c + 16);
    aq[1].u[0] = *(const v4u*)(qb + 32 + c);
    aq[1].u[1] = *(const v4u*)(qb + 32 + c + 16);

    v8f oacc[4] = {};
    float rmax[8], rsum[8];
#pragma unroll
    for (int v = 0; v < 8; ++v) { rmax[v] = -1e30f; rsum[v] = 0.0f; }

    for (int kt = 0; kt < Sc / 64; ++kt) {
        // ---- scores: S = Q @ K^T  (K rows are exactly B-fragment columns) ----
        v8f sacc[4] = {};
#pragma unroll
        for (int nt = 0; nt < 4; ++nt) {
            int key = kt * 64 + nt * 16 + m16;
            const u16* kb = kh + (bh * Sc + key) * DHc;
            Frag bk0, bk1;
            bk0.u[0] = *(const v4u*)(kb + c);       bk0.u[1] = *(const v4u*)(kb + c + 16);
            bk1.u[0] = *(const v4u*)(kb + 32 + c);  bk1.u[1] = *(const v4u*)(kb + 32 + c + 16);
            sacc[nt] = WMMA_BF16(aq[0].v, bk0.v, sacc[nt]);
            sacc[nt] = WMMA_BF16(aq[1].v, bk1.v, sacc[nt]);
            float biasv = (mask[b * Sc + key] > 0) ? 0.0f : -3.4028235e38f;
#pragma unroll
            for (int v = 0; v < 8; ++v) sacc[nt][v] += biasv;
        }

        // ---- online softmax: row m = v + 8*lh, replicated across 16 lanes ----
#pragma unroll
        for (int v = 0; v < 8; ++v) {
            float mx = fmaxf(fmaxf(sacc[0][v], sacc[1][v]), fmaxf(sacc[2][v], sacc[3][v]));
#pragma unroll
            for (int off = 8; off >= 1; off >>= 1) mx = fmaxf(mx, __shfl_xor(mx, off, 32));
            float mnew  = fmaxf(rmax[v], mx);
            float alpha = __expf(rmax[v] - mnew);
            rmax[v] = mnew;
            float ps = 0.0f;
#pragma unroll
            for (int nt = 0; nt < 4; ++nt) {
                float p = __expf(sacc[nt][v] - mnew);
                sacc[nt][v] = p;
                ps += p;
            }
#pragma unroll
            for (int off = 8; off >= 1; off >>= 1) ps += __shfl_xor(ps, off, 32);
            rsum[v] = rsum[v] * alpha + ps;
#pragma unroll
            for (int nt = 0; nt < 4; ++nt) oacc[nt][v] *= alpha;
        }

        // ---- transpose P through per-wave LDS (same-wave DS ops are in-order) ----
#pragma unroll
        for (int v = 0; v < 8; ++v) {
            int m = v + 8 * lh;
#pragma unroll
            for (int nt = 0; nt < 4; ++nt)
                pP[w][m][nt * 16 + m16] = f2bf(sacc[nt][v]);
        }

        // ---- O += P @ V  (vt is [dh][s]: contiguous B-fragment columns) ----
#pragma unroll
        for (int j = 0; j < 2; ++j) {
            const u16* pb = &pP[w][m16][j * 32];
            Frag ap;
            ap.u[0] = *(const v4u*)(pb + c);
            ap.u[1] = *(const v4u*)(pb + c + 16);
#pragma unroll
            for (int nt = 0; nt < 4; ++nt) {
                int dh = nt * 16 + m16;
                const u16* vb = vt + (bh * DHc + dh) * Sc + kt * 64 + j * 32;
                Frag bv;
                bv.u[0] = *(const v4u*)(vb + c);
                bv.u[1] = *(const v4u*)(vb + c + 16);
                oacc[nt] = WMMA_BF16(ap.v, bv.v, oacc[nt]);
            }
        }
    }

    // ---- epilogue: normalize, head mask, write [b,s,h*64+dh] f32 ----
    float hmv = hm[h];
#pragma unroll
    for (int v = 0; v < 8; ++v) {
        float sc2 = hmv / rsum[v];
        int q = qt * 64 + w * 16 + v + 8 * lh;
#pragma unroll
        for (int nt = 0; nt < 4; ++nt) {
            int dh = nt * 16 + m16;
            out[((size_t)b * Sc + q) * Dc + h * DHc + dh] = oacc[nt][v] * sc2;
        }
    }
}

// ---------------------------------------------------------------------------
// Host launcher
// ---------------------------------------------------------------------------
extern "C" void kernel_launch(void* const* d_in, const int* in_sizes, int n_in,
                              void* d_out, int out_size, void* d_ws, size_t ws_size,
                              hipStream_t stream) {
    const float* hs   = (const float*)d_in[0];
    const int*   amsk = (const int*)  d_in[1];
    const float* spos = (const float*)d_in[2];
    const float* hmsk = (const float*)d_in[3];
    const float* Wq   = (const float*)d_in[4];
    const float* bq   = (const float*)d_in[5];
    const float* Wk   = (const float*)d_in[6];
    const float* bk   = (const float*)d_in[7];
    const float* Wv   = (const float*)d_in[8];
    const float* bv   = (const float*)d_in[9];
    float* out = (float*)d_out;

    char* w = (char*)d_ws;
    u16*   hsb  = (u16*)  (w);                        //  8 MB bf16 hidden states
    u16*   wqb  = (u16*)  (w + ((size_t) 8 << 20));   //  2 MB (transposed [N][K])
    u16*   wkb  = (u16*)  (w + ((size_t)10 << 20));   //  2 MB
    u16*   wvb  = (u16*)  (w + ((size_t)12 << 20));   //  2 MB
    float* qraw = (float*)(w + ((size_t)16 << 20));   // 16 MB
    float* kraw = (float*)(w + ((size_t)32 << 20));   // 16 MB
    float* vraw = (float*)(w + ((size_t)48 << 20));   // 16 MB
    u16*   qhp  = (u16*)  (w + ((size_t)64 << 20));   //  8 MB
    u16*   khp  = (u16*)  (w + ((size_t)72 << 20));   //  8 MB
    u16*   vtp  = (u16*)  (w + ((size_t)80 << 20));   //  8 MB

    const int nHS = Bc * Sc * Dc;   // 4194304

    cvt_bf16_kernel<<<nHS / 256, 256, 0, stream>>>(hs, hsb, nHS);

    dim3 tgrid(Dc / 32, Dc / 32);                     // (32, 32)
    cvt_transpose_kernel<<<tgrid, 256, 0, stream>>>(Wq, wqb);
    cvt_transpose_kernel<<<tgrid, 256, 0, stream>>>(Wk, wkb);
    cvt_transpose_kernel<<<tgrid, 256, 0, stream>>>(Wv, wvb);

    dim3 ggrid(Dc / 128, (Bc * Sc) / 128, 3);         // (8, 32, 3)
    qkv_gemm_kernel<<<ggrid, 256, 0, stream>>>(hsb, wqb, wkb, wvb, bq, bk, bv,
                                               qraw, kraw, vraw);

    int nRope = Bc * Sc * Hc * 32;                    // 2097152
    rope_pack_kernel<<<nRope / 256, 256, 0, stream>>>(qraw, kraw, vraw, spos,
                                                      qhp, khp, vtp);

    dim3 agrid(Sc / 64, Hc, Bc);                      // (32, 16, 2)
    attn_kernel<<<agrid, 128, 0, stream>>>(qhp, khp, vtp, amsk, hmsk, out);
}
